// HeavyNet_37830071943760
// MI455X (gfx1250) — matrix operand order repaired and checked by reading, not compiled, not run on hardware
//
#include <hip/hip_runtime.h>

typedef __attribute__((ext_vector_type(16))) __bf16 v16bf;
typedef __attribute__((ext_vector_type(8)))  __bf16 v8bf;
typedef __attribute__((ext_vector_type(8)))  float  v8f;

#define NN_NODES 50000
#define CH       256
#define NN_EDGES 800000
#define NN_LABEL 100000

// ---------------------------------------------------------------- utilities
__global__ void cvt_bf16_kernel(const float* __restrict__ src,
                                __bf16* __restrict__ dst, int n) {
  int i = blockIdx.x * blockDim.x + threadIdx.x;
  if (i < n) dst[i] = (__bf16)src[i];
}

// Repack W [K,N] f32 row-major into WMMA B-fragment order:
// tile (kT,nT) of 32x16; lane = half*16 + (n%16); lane's 16 bf16 contiguous,
// element s <-> K = kT*32 + half*8 + s (s<8), K = kT*32 + 16 + half*8 + s-8 (s>=8).
__global__ void pack_w_kernel(const float* __restrict__ W,
                              __bf16* __restrict__ out, int K, int N) {
  int i = blockIdx.x * blockDim.x + threadIdx.x;
  if (i >= K * N) return;
  int k = i / N, n = i - k * N;
  int kT = k >> 5, kr = k & 31;
  int half = (kr >> 3) & 1, v = ((kr & 7) | ((kr >> 4) << 3)) >> 1, lo = kr & 1;
  int nT = n >> 4, nloc = n & 15;
  int lane = (half << 4) | nloc;
  size_t idx = ((size_t)(kT * (N >> 4) + nT) * 32 + lane) * 16 + 2 * v + lo;
  out[idx] = (__bf16)W[i];
}

__global__ void fill_kernel(float* __restrict__ p, float v, int n) {
  int i = blockIdx.x * blockDim.x + threadIdx.x;
  if (i < n) p[i] = v;
}

__global__ void deg_kernel(const int* __restrict__ dst, float* __restrict__ deg, int E) {
  int e = blockIdx.x * blockDim.x + threadIdx.x;
  if (e < E) atomicAdd(&deg[dst[e]], 1.0f);
}

__global__ void rsqrt_kernel(float* __restrict__ d, int n) {
  int i = blockIdx.x * blockDim.x + threadIdx.x;
  if (i < n) {
    float v = d[i];
    d[i] = (v > 0.0f) ? rsqrtf(fmaxf(v, 1e-12f)) : 0.0f;
  }
}

__global__ void relu_cvt_kernel(const float* __restrict__ src,
                                __bf16* __restrict__ dst, int n) {
  int i = blockIdx.x * blockDim.x + threadIdx.x;
  if (i < n) dst[i] = (__bf16)fmaxf(src[i], 0.0f);
}

// agg[n,c] = h[n,c]*dinv[n]^2 + bias[c]   (self-loop contribution + bias)
__global__ void agg_init_kernel(const float* __restrict__ h,
                                const float* __restrict__ dinv,
                                const float* __restrict__ bias,
                                float* __restrict__ agg, int nNodes) {
  int i = blockIdx.x * blockDim.x + threadIdx.x;
  if (i < nNodes * CH) {
    int node = i >> 8, c = i & 255;
    float d = dinv[node];
    agg[i] = h[i] * d * d + bias[c];
  }
}

// per-edge scatter: agg[dst] += h[src] * dinv[src]*dinv[dst]; 64 lanes/edge, float4
__global__ __launch_bounds__(256)
void scatter_kernel(const int* __restrict__ src, const int* __restrict__ dst,
                    const float* __restrict__ h, const float* __restrict__ dinv,
                    float* __restrict__ agg, int E) {
  int e = blockIdx.x * 4 + (threadIdx.x >> 6);
  if (e >= E) return;
  int s = src[e], d = dst[e];
  float nrm = dinv[s] * dinv[d];
  int c = (threadIdx.x & 63) * 4;
  const float4 hv = *(const float4*)(h + (size_t)s * CH + c);
  float* ap = agg + (size_t)d * CH + c;
  atomicAdd(ap + 0, hv.x * nrm);
  atomicAdd(ap + 1, hv.y * nrm);
  atomicAdd(ap + 2, hv.z * nrm);
  atomicAdd(ap + 3, hv.w * nrm);
}

// A-fragment (16-bit 16x32 layout) = two contiguous 16B chunks of the row:
// elems 0..7 <-> K = kb + half*8 + 0..7 ; elems 8..15 <-> K = kb+16+half*8+0..7
static __device__ __forceinline__ v16bf load_a_frag(const __bf16* rowbase,
                                                    int kb, int half) {
  const __bf16* ap = rowbase + kb + (half << 3);
  v8bf lo = *(const v8bf*)ap;
  v8bf hi = *(const v8bf*)(ap + 16);
  return __builtin_shufflevector(lo, hi, 0, 1, 2, 3, 4, 5, 6, 7,
                                 8, 9, 10, 11, 12, 13, 14, 15);
}

// ------------------------------------------------- WMMA GEMM: C[M,256]=A@B
// A: bf16 [M,256] row-major; Bp: packed fragment order (pack_w_kernel, N=256)
// grid (M/16, 2), block 256 (8 waves); wave w -> 16x16 tile at col (by*8+w)*16
__global__ __launch_bounds__(256)
void gemm_bf16_256(const __bf16* __restrict__ A, const __bf16* __restrict__ Bp,
                   float* __restrict__ C, int M) {
  __shared__ __bf16 sA[16 * 264];             // 16 rows x 256, padded stride 264
  const int mrow = blockIdx.x * 16;
  const int t = threadIdx.x;

  { // cooperative A-tile load: 32 uint4 per row, 2 per thread
    const uint4* g = (const uint4*)(A + (size_t)mrow * 256);
    uint4* s = (uint4*)sA;
    int row = t >> 4, kv = (t & 15) * 2;
    s[row * 33 + kv]     = g[row * 32 + kv];
    s[row * 33 + kv + 1] = g[row * 32 + kv + 1];
  }
  __syncthreads();

  const int wave = t >> 5, lane = t & 31;
  const int half = lane >> 4, nloc = lane & 15, mr = lane & 15;
  const int nT = blockIdx.y * 8 + wave;       // column tile index (16 cols)
  const __bf16* rowbase = &sA[mr * 264];
  const v16bf* B16 = (const v16bf*)Bp;

  v8f acc = {};
#pragma unroll
  for (int kb = 0; kb < 256; kb += 32) {
    v16bf a = load_a_frag(rowbase, kb, half);
    v16bf b = B16[(size_t)((kb >> 5) * 16 + nT) * 32 + lane];
    acc = __builtin_amdgcn_wmma_f32_16x16x32_bf16(
        false, a, false, b, (short)0, acc, false, false);
  }

  float* cp = C + (size_t)(mrow + (half << 3)) * 256 + nT * 16 + nloc;
#pragma unroll
  for (int r = 0; r < 8; ++r) cp[(size_t)r * 256] = acc[r];
}

// ------------------------------------------------------- fused decoder
// out[e] = bm2 + sum_c relu(pair[e]·Wm1[:,c] + bm1[c]) * Wm2[c]
// pair = [zi, zj, |zi-zj|] built on the fly from LDS-staged z rows.
// grid (N_LABEL/16), block 512 (16 waves, wave w -> channels w*16..w*16+15)
__global__ void decode_init_kernel(float* __restrict__ out,
                                   const float* __restrict__ bm2, int n) {
  int i = blockIdx.x * blockDim.x + threadIdx.x;
  if (i < n) out[i] = bm2[0];
}

// |a-b| with the sign-clear done in opaque inline asm. Any C-level abs
// (fabsf, maxnum(x,-x), select, bit-twiddle) is canonicalized by InstCombine
// into llvm.fabs, which this gfx1250 backend miscombines into an unselectable
// v2f32-bitcast-and node when SLP-vectorized. v_and_b32 with 0x7fffffff is
// exactly what v_fabs costs, but invisible to the optimizer.
static __device__ __forceinline__ float absdiff(float a, float b) {
  float d = a - b;
  float r;
  asm("v_and_b32 %0, 0x7fffffff, %1" : "=v"(r) : "v"(d));
  return r;
}

__global__ __launch_bounds__(512)
void decode_kernel(const float* __restrict__ z,
                   const int* __restrict__ eli0, const int* __restrict__ eli1,
                   const __bf16* __restrict__ Wm1p, const float* __restrict__ bm1,
                   const float* __restrict__ wm2, float* __restrict__ out) {
  __shared__ float zl[16][516];               // cols 0..255 = zi, 256..511 = zj
  const int et = blockIdx.x;
  const int t = threadIdx.x;

  // stage 16 edges x (zi,zj): 2048 float4, 4 per thread, coalesced
#pragma unroll
  for (int j = 0; j < 4; ++j) {
    int fid = j * 512 + t;
    int row = fid >> 7, colv = fid & 127;
    int idx = (colv < 64) ? eli0[et * 16 + row] : eli1[et * 16 + row];
    int cv = colv & 63;
    float4 v = ((const float4*)(z + (size_t)idx * 256))[cv];
    *(float4*)&zl[row][colv * 4] = v;
  }
  __syncthreads();

  const int wave = t >> 5, lane = t & 31;
  const int half = lane >> 4, nloc = lane & 15, mr = lane & 15;
  const float* rowp = &zl[mr][0];
  const v16bf* B16 = (const v16bf*)Wm1p;

  v8f acc = {};
#pragma unroll
  for (int kb = 0; kb < 768; kb += 32) {      // kb constant per iter: region
    v16bf a;                                   // branch folds at compile time
#pragma unroll
    for (int v = 0; v < 8; ++v) {
      int kl = 2 * (v & 3) + ((v >> 2) << 4) + (half << 3);
      int k = kb + kl;
      float a0, a1;
      if (kb < 512) {                          // zi or zj region
        a0 = rowp[k];
        a1 = rowp[k + 1];
      } else {                                 // |zi - zj| region
        a0 = absdiff(rowp[k - 512], rowp[k - 256]);
        a1 = absdiff(rowp[k - 511], rowp[k - 255]);
      }
      a[2 * v]     = (__bf16)a0;
      a[2 * v + 1] = (__bf16)a1;
    }
    v16bf b = B16[(size_t)((kb >> 5) * 16 + wave) * 32 + lane];
    acc = __builtin_amdgcn_wmma_f32_16x16x32_bf16(
        false, a, false, b, (short)0, acc, false, false);
  }

  const float bias = bm1[wave * 16 + nloc];
  const float w2v  = wm2[wave * 16 + nloc];
#pragma unroll
  for (int r = 0; r < 8; ++r) {
    float p = fmaxf(acc[r] + bias, 0.0f) * w2v;
    p += __shfl_xor(p, 1, 16);
    p += __shfl_xor(p, 2, 16);
    p += __shfl_xor(p, 4, 16);
    p += __shfl_xor(p, 8, 16);
    if (nloc == 0)
      atomicAdd(&out[et * 16 + (half << 3) + r], p);
  }
}

// ---------------------------------------------------------------- launcher
extern "C" void kernel_launch(void* const* d_in, const int* in_sizes, int n_in,
                              void* d_out, int out_size, void* d_ws, size_t ws_size,
                              hipStream_t stream) {
  const float* x   = (const float*)d_in[0];
  const int*   ei  = (const int*)d_in[1];     // [2, E]: src row 0, dst row 1
  const int*   eli = (const int*)d_in[2];     // [2, L]
  const float* W1  = (const float*)d_in[3];
  const float* b1  = (const float*)d_in[4];
  const float* W2  = (const float*)d_in[5];
  const float* b2  = (const float*)d_in[6];
  const float* Wm1 = (const float*)d_in[7];
  const float* bm1 = (const float*)d_in[8];
  const float* Wm2 = (const float*)d_in[9];
  const float* bm2 = (const float*)d_in[10];
  float* out = (float*)d_out;

  char* ws = (char*)d_ws;
  size_t off = 0;
  auto take = [&](size_t bytes) -> char* {
    char* p = ws + off;
    off += (bytes + 1023) & ~(size_t)1023;
    return p;
  };
  __bf16* xb   = (__bf16*)take((size_t)NN_NODES * CH * 2);
  __bf16* w1p  = (__bf16*)take((size_t)CH * CH * 2);
  __bf16* w2p  = (__bf16*)take((size_t)CH * CH * 2);
  __bf16* wm1p = (__bf16*)take((size_t)3 * CH * CH * 2);
  float*  t0   = (float*)take((size_t)NN_NODES * CH * 4);   // gemm output
  float*  agg1 = (float*)take((size_t)NN_NODES * CH * 4);   // conv1 aggregate
  __bf16* z1b  = (__bf16*)take((size_t)NN_NODES * CH * 2);  // relu(conv1) bf16
  float*  zf   = (float*)take((size_t)NN_NODES * CH * 4);   // conv2 out = z
  float*  dinv = (float*)take((size_t)NN_NODES * 4);        // deg then d^-1/2

  const int nxc = NN_NODES * CH;
  const int* esrc = ei;
  const int* edst = ei + NN_EDGES;
  const int* li0  = eli;
  const int* li1  = eli + NN_LABEL;

  // bf16 conversion (x) + weight repack into WMMA fragment order
  cvt_bf16_kernel<<<(nxc + 255) / 256, 256, 0, stream>>>(x, xb, nxc);
  pack_w_kernel<<<(CH * CH + 255) / 256, 256, 0, stream>>>(W1, w1p, CH, CH);
  pack_w_kernel<<<(CH * CH + 255) / 256, 256, 0, stream>>>(W2, w2p, CH, CH);
  pack_w_kernel<<<(3 * CH * CH + 255) / 256, 256, 0, stream>>>(Wm1, wm1p, 3 * CH, CH);

  // degrees (self-loop => init 1.0) and d^-1/2
  fill_kernel<<<(NN_NODES + 255) / 256, 256, 0, stream>>>(dinv, 1.0f, NN_NODES);
  deg_kernel<<<(NN_EDGES + 255) / 256, 256, 0, stream>>>(edst, dinv, NN_EDGES);
  rsqrt_kernel<<<(NN_NODES + 255) / 256, 256, 0, stream>>>(dinv, NN_NODES);

  dim3 ggrid(NN_NODES / 16, 2);

  // conv1: t0 = x@W1 ; agg1 = self-loop+bias ; scatter ; relu->bf16
  gemm_bf16_256<<<ggrid, 256, 0, stream>>>(xb, w1p, t0, NN_NODES);
  agg_init_kernel<<<(nxc + 255) / 256, 256, 0, stream>>>(t0, dinv, b1, agg1, NN_NODES);
  scatter_kernel<<<NN_EDGES / 4, 256, 0, stream>>>(esrc, edst, t0, dinv, agg1, NN_EDGES);
  relu_cvt_kernel<<<(nxc + 255) / 256, 256, 0, stream>>>(agg1, z1b, nxc);

  // conv2: t0 = relu(z1)@W2 ; zf = self-loop+bias ; scatter (no relu)
  gemm_bf16_256<<<ggrid, 256, 0, stream>>>(z1b, w2p, t0, NN_NODES);
  agg_init_kernel<<<(nxc + 255) / 256, 256, 0, stream>>>(t0, dinv, b2, zf, NN_NODES);
  scatter_kernel<<<NN_EDGES / 4, 256, 0, stream>>>(esrc, edst, t0, dinv, zf, NN_EDGES);

  // fused decoder
  decode_init_kernel<<<(NN_LABEL + 255) / 256, 256, 0, stream>>>(out, bm2, NN_LABEL);
  decode_kernel<<<NN_LABEL / 16, 512, 0, stream>>>(zf, li0, li1, wm1p, bm1, Wm2, out);
}
